// BitStackLinear_60455959658632
// MI455X (gfx1250) — matrix operand order repaired and checked by reading, not compiled, not run on hardware
//
#include <hip/hip_runtime.h>

typedef __attribute__((ext_vector_type(16))) _Float16 v16h;
typedef __attribute__((ext_vector_type(8)))  _Float16 v8h;
typedef __attribute__((ext_vector_type(8)))  float    v8f;
typedef __attribute__((ext_vector_type(4)))  int      v4i;

#define IN_F   4096
#define OUT_F  4096
#define W_BITS 4
#define RANK   16
#define M_TOT  8192   // B * S = 4 * 2048

// ---------------------------------------------------------------------------
// CDNA5 async global->LDS copy (ASYNCcnt-tracked, no VGPR staging), guarded.
// Builtin signature (probe-confirmed): (v4i AS1*, v4i AS3*, imm, imm).
// ---------------------------------------------------------------------------
#if defined(__gfx1250__) && __has_builtin(__builtin_amdgcn_global_load_async_to_lds_b128)
#define HAS_ASYNC_LDS 1
typedef __attribute__((address_space(1))) v4i as1_v4i;
typedef __attribute__((address_space(3))) v4i as3_v4i;
#else
#define HAS_ASYNC_LDS 0
#endif

__device__ __forceinline__ void async_copy16(const _Float16* g, _Float16* l) {
#if HAS_ASYNC_LDS
    __builtin_amdgcn_global_load_async_to_lds_b128(
        (as1_v4i*)g, (as3_v4i*)l, 0, 0);
#else
    *(uint4*)l = *(const uint4*)g;
#endif
}

template <int N>
__device__ __forceinline__ void wait_async() {
#if HAS_ASYNC_LDS
#if __has_builtin(__builtin_amdgcn_s_wait_asynccnt)
    __builtin_amdgcn_s_wait_asynccnt(N);
#else
    asm volatile("s_wait_asynccnt %0" ::"i"(N) : "memory");
#endif
#endif
}

// ---------------------------------------------------------------------------
// Kernel 1: materialize W[o,i] = sum_b sign_b(o,i) * (u_b @ vt_b)[o,i]  (fp16)
// Tile: 32 (out) x 256 (in) per block, 256 threads.
// ---------------------------------------------------------------------------
#define OT 32
#define IT 256

__global__ __launch_bounds__(256) void build_w_kernel(
    const int* __restrict__ qw,        // [4][OUT_F*IN_F/8] (one byte per int32)
    const _Float16* __restrict__ u,    // [4][OUT_F][16]
    const _Float16* __restrict__ vt,   // [4][16][IN_F]
    _Float16* __restrict__ W)          // [OUT_F][IN_F]
{
    __shared__ _Float16 uS[W_BITS * OT * RANK];   // 4 KB
    __shared__ _Float16 vS[W_BITS * RANK * IT];   // 32 KB

    const int t  = threadIdx.x;
    const int i0 = blockIdx.x * IT;
    const int o0 = blockIdx.y * OT;

    {
        int b  = t >> 6;
        int ol = (t >> 1) & 31;
        int kh = (t & 1) * 8;
        uint4 d = *(const uint4*)(u + ((size_t)b * OUT_F + (o0 + ol)) * RANK + kh);
        *(uint4*)(uS + (b * OT + ol) * RANK + kh) = d;
    }
#pragma unroll
    for (int j = 0; j < 8; ++j) {
        int c   = t + j * 256;
        int row = c >> 5;             // 0..63 ((b,k) pair)
        int ii  = (c & 31) * 8;
        int b   = row >> 4;
        int k   = row & 15;
        uint4 d = *(const uint4*)(vt + ((size_t)(b * RANK + k)) * IN_F + i0 + ii);
        *(uint4*)(vS + (b * RANK + k) * IT + ii) = d;
    }
    __syncthreads();

    const int ol = t >> 3;
    const int ti = t & 7;
    const int og = o0 + ol;

    float ur[W_BITS][RANK];
#pragma unroll
    for (int b = 0; b < W_BITS; ++b)
#pragma unroll
        for (int k = 0; k < RANK; ++k)
            ur[b][k] = (float)uS[(b * OT + ol) * RANK + k];

#pragma unroll
    for (int j = 0; j < 4; ++j) {
        const int ib = ti * 8 + j * 64;
        float acc[8];
#pragma unroll
        for (int e = 0; e < 8; ++e) acc[e] = 0.0f;

#pragma unroll
        for (int b = 0; b < W_BITS; ++b) {
            float lr[8];
#pragma unroll
            for (int e = 0; e < 8; ++e) lr[e] = 0.0f;
#pragma unroll
            for (int k = 0; k < RANK; ++k) {
                v8h vv = *(const v8h*)(vS + (b * RANK + k) * IT + ib);
                float us = ur[b][k];
#pragma unroll
                for (int e = 0; e < 8; ++e) lr[e] += us * (float)vv[e];
            }
            int q = qw[(size_t)b * (OUT_F * (IN_F / 8)) + (size_t)og * (IN_F / 8)
                       + ((i0 + ib) >> 3)];
#pragma unroll
            for (int e = 0; e < 8; ++e)
                acc[e] += ((q >> e) & 1) ? lr[e] : -lr[e];
        }

        v8h r;
#pragma unroll
        for (int e = 0; e < 8; ++e) r[e] = (_Float16)acc[e];
        *(v8h*)(W + (size_t)og * IN_F + i0 + ib) = r;
    }
}

// ---------------------------------------------------------------------------
// Kernel 2: Out[m,n] = sum_k X[m,k] * W[n,k]
// Block tile 128x256, BK=32, 256 threads = 8 waves (2 M x 4 N),
// each wave: 64x64 region = 4x4 v_wmma_f32_16x16x32_f16 fragments.
// Triple-buffered LDS fed by async global->LDS loads (2 tiles in flight);
// coalesced epilogue through an LDS stage (reuses the K-loop buffers).
// ---------------------------------------------------------------------------
#define BM 128
#define BN 256
#define BK 32
#define LDR 40        // halfs per LDS row (32 + 8 pad): 80B stride, 16B aligned
#define NT (IN_F / BK)
#define STR 264       // epilogue stage stride in halfs (256 + 8 pad)

__global__ __launch_bounds__(256) void gemm_kernel(
    const _Float16* __restrict__ X,    // [M_TOT][IN_F]
    const _Float16* __restrict__ W,    // [OUT_F][IN_F]
    _Float16* __restrict__ Out)        // [M_TOT][OUT_F]
{
    __shared__ union {
        struct {
            _Float16 a[3][BM * LDR];   // 3 x 10 KB
            _Float16 b[3][BN * LDR];   // 3 x 20 KB
        } kbuf;                        // 90 KB
        _Float16 stage[BM * STR];      // 66 KB epilogue stage
    } sh;

    const int t    = threadIdx.x;
    const int gN   = blockIdx.x * BN;
    const int gM   = blockIdx.y * BM;
    const int wave = t >> 5;
    const int lane = t & 31;
    const int wm   = (wave >> 2) * 64;   // wave M offset (0,64)
    const int wn   = (wave & 3) * 64;    // wave N offset (0..192)
    const int mrow = lane & 15;
    const int half = lane >> 4;          // lane group 0/1

    v8f acc[4][4];
#pragma unroll
    for (int fi = 0; fi < 4; ++fi)
#pragma unroll
        for (int fj = 0; fj < 4; ++fj)
#pragma unroll
            for (int e = 0; e < 8; ++e) acc[fi][fj][e] = 0.0f;

    // Per-thread staging chunks: A = 512 16B chunks, B = 1024 16B chunks.
    const int arow = t >> 2;             // 0..63 base row
    const int aco  = (t & 3) * 8;        // half offset within 32-wide k slab

    const _Float16* xbase = X + (size_t)gM * IN_F;
    const _Float16* wbase = W + (size_t)gN * IN_F;

    // Stage one K-tile (6 async b128 loads per thread) into buffer `buf`.
    auto stage_tile = [&](int buf, int kk) {
#pragma unroll
        for (int i = 0; i < 2; ++i) {
            int row = arow + i * 64;
            async_copy16(xbase + (size_t)row * IN_F + kk + aco,
                         &sh.kbuf.a[buf][row * LDR + aco]);
        }
#pragma unroll
        for (int i = 0; i < 4; ++i) {
            int row = arow + i * 64;
            async_copy16(wbase + (size_t)row * IN_F + kk + aco,
                         &sh.kbuf.b[buf][row * LDR + aco]);
        }
    };

    // Prologue: two tiles in flight.
    stage_tile(0, 0);
    stage_tile(1, BK);

    int cur = 0;
    for (int ti = 0; ti < NT; ++ti) {
        // Drain only the oldest tile (in-order completion): keep the next
        // tile's 6 loads outstanding, except at the last iteration.
        if (ti + 1 < NT) wait_async<6>();
        else             wait_async<0>();
        __syncthreads();   // tile `ti` visible to all; buf (cur+2)%3 free

        if (ti + 2 < NT) {
            int nxt2 = cur + 2; if (nxt2 >= 3) nxt2 -= 3;
            stage_tile(nxt2, (ti + 2) * BK);
        }

        // Load all 8 fragments, then issue the 4x4 WMMA outer product.
        v16h af[4], bf[4];
#pragma unroll
        for (int fi = 0; fi < 4; ++fi) {
            const _Float16* p = &sh.kbuf.a[cur][(wm + fi * 16 + mrow) * LDR + half * 8];
            v8h lo = *(const v8h*)p;          // K = half*8 + 0..7
            v8h hi = *(const v8h*)(p + 16);   // K = 16 + half*8 + 0..7
            af[fi] = __builtin_shufflevector(lo, hi, 0, 1, 2, 3, 4, 5, 6, 7,
                                             8, 9, 10, 11, 12, 13, 14, 15);
        }
#pragma unroll
        for (int fj = 0; fj < 4; ++fj) {
            const _Float16* p = &sh.kbuf.b[cur][(wn + fj * 16 + mrow) * LDR + half * 8];
            v8h lo = *(const v8h*)p;          // K = half*8 + 0..7
            v8h hi = *(const v8h*)(p + 16);   // K = 16 + half*8 + 0..7
            bf[fj] = __builtin_shufflevector(lo, hi, 0, 1, 2, 3, 4, 5, 6, 7,
                                             8, 9, 10, 11, 12, 13, 14, 15);
        }
#pragma unroll
        for (int fi = 0; fi < 4; ++fi)
#pragma unroll
            for (int fj = 0; fj < 4; ++fj)
                acc[fi][fj] = __builtin_amdgcn_wmma_f32_16x16x32_f16(
                    false, af[fi], false, bf[fj], (short)0, acc[fi][fj],
                    false, false);

        ++cur; if (cur >= 3) cur = 0;
    }

    // ---------------- Coalesced epilogue through LDS stage ----------------
    __syncthreads();   // everyone done with kbuf before aliasing it as stage
#pragma unroll
    for (int fi = 0; fi < 4; ++fi)
#pragma unroll
        for (int fj = 0; fj < 4; ++fj) {
            // C/D layout: m = v + 8*half, n = lane%16 within the fragment.
            _Float16* p = &sh.stage[(wm + fi * 16 + half * 8) * STR
                                    + wn + fj * 16 + mrow];
#pragma unroll
            for (int v = 0; v < 8; ++v)
                p[v * STR] = (_Float16)acc[fi][fj][v];
        }
    __syncthreads();

    // 128 x 128 (cols 0..127) of the tile -> 2048 b128 chunks -> 8 per thread, fully coalesced.
#pragma unroll
    for (int i = 0; i < 8; ++i) {
        int c   = t + i * 256;       // 0..2047
        int row = c >> 4;            // 0..127
        int co  = (c & 15) * 8;      // 0..120
        uint4 d = *(const uint4*)(&sh.stage[row * STR + co]);
        *(uint4*)(Out + (size_t)(gM + row) * OUT_F + gN + co) = d;
    }
}

// ---------------------------------------------------------------------------
extern "C" void kernel_launch(void* const* d_in, const int* in_sizes, int n_in,
                              void* d_out, int out_size, void* d_ws, size_t ws_size,
                              hipStream_t stream) {
    const _Float16* x  = (const _Float16*)d_in[0];  // [4][2048][4096] fp16
    const int*      qw = (const int*)d_in[1];       // [4][2097152] int32
    const _Float16* u  = (const _Float16*)d_in[2];  // [4][4096][16] fp16
    const _Float16* vt = (const _Float16*)d_in[3];  // [4][16][4096] fp16
    _Float16* out = (_Float16*)d_out;               // [4][2048][4096] fp16
    _Float16* W   = (_Float16*)d_ws;                // [4096][4096] fp16 = 32 MiB

    (void)in_sizes; (void)n_in; (void)out_size; (void)ws_size;

    build_w_kernel<<<dim3(IN_F / IT, OUT_F / OT), 256, 0, stream>>>(qw, u, vt, W);
    gemm_kernel<<<dim3(OUT_F / BN, M_TOT / BM), 256, 0, stream>>>(x, W, out);
}
